// Block_Base_15212774162754
// MI455X (gfx1250) — matrix operand order repaired and checked by reading, not compiled, not run on hardware
//
#include <hip/hip_runtime.h>

// ---------------------------------------------------------------------------
// MI455X (gfx1250) fused masked-MHA: bf16 WMMA GEMMs + flash attention,
// with async global->LDS staging (ASYNCcnt) and global prefetch.
// ---------------------------------------------------------------------------

typedef __bf16 bf16_t;
typedef __attribute__((ext_vector_type(16))) __bf16 v16bf;
typedef __attribute__((ext_vector_type(8)))  __bf16 v8bf;
typedef __attribute__((ext_vector_type(8)))  float  v8f;
typedef __attribute__((ext_vector_type(4)))  float  v4f;

// Problem constants (setup_inputs is deterministic).
constexpr int kB   = 2;
constexpr int kT   = 1024;
constexpr int kN   = 64;
constexpr int kC   = 512;
constexpr int kH   = 8;
constexpr int kDH  = 64;
constexpr int kMot = 3 * kT;        // 3072
constexpr int kL   = kMot + 5 * kN; // 3392
constexpr int kM   = kB * kL;       // 6784

#define DEVFN static __device__ __forceinline__

DEVFN v8f wmma_bf16(v16bf a, v16bf b, v8f c) {
  // v_wmma_f32_16x16x32_bf16  D = A(16x32) * B(32x16) + C
  return __builtin_amdgcn_wmma_f32_16x16x32_bf16(false, a, false, b,
                                                 (short)0, c, false, false);
}

DEVFN v16bf cat8(v8bf lo, v8bf hi) {
  v16bf r;
#pragma unroll
  for (int e = 0; e < 8; ++e) { r[e] = lo[e]; r[e + 8] = hi[e]; }
  return r;
}

// Async DMA-style copy: 16 bytes per lane, global -> LDS, tracked on ASYNCcnt.
// VDST operand carries the per-lane LDS byte address (low 32 bits of the flat
// LDS-aperture address), VADDR the 64-bit global address (GV mode, saddr=off).
DEVFN void async_load_b128_to_lds(const void* gptr, void* lptr) {
  asm volatile("global_load_async_to_lds_b128 %0, %1, off"
               :: "v"((unsigned)(size_t)lptr),
                  "v"((unsigned long long)(size_t)gptr)
               : "memory");
}
DEVFN void wait_asynccnt0() {
  asm volatile("s_wait_asynccnt 0" ::: "memory");
}

// ---------------------------------------------------------------------------
// Structured mask (mirrors build_mask): true = attention allowed.
// ---------------------------------------------------------------------------
DEVFN bool allowed_rc(int r, int c) {
  if (r < kMot) {                       // motion query
    const int i = r / kT, rr = r % kT;
    if (c < kMot) {                     // motion key: block-causal structure
      const int j = c / kT, cc = c % kT;
      if (i == 0) return (j == 0) && (cc <= rr);
      if (i == 1) return (j == 0) ? (cc <= rr) : (cc < rr);
      return (j == 2) ? (cc < rr) : (cc <= rr);   // i == 2
    } else {                            // motion -> text (part_only carve)
      const int d = (c - kMot) / kN;
      if (i == 0) return false;
      if (i == 1) return (d == 0) || (d >= 2);    // upper body
      return d >= 1;                               // lower body
    }
  }
  return c >= kMot;                     // text query: text keys only
}

// Tile classification for a 16-row query tile x 32-col key tile.
// 0 = fully blocked (skip), 1 = fully allowed, 2 = partial (elementwise mask).
DEVFN int tile_status(int qr0, int kc0) {
  if (qr0 >= kMot) return (kc0 >= kMot) ? 1 : 0;       // text query rows
  const int i = qr0 / kT, rr0 = qr0 % kT, rr1 = rr0 + 15;
  if (kc0 >= kMot) {                                   // motion -> text
    const int d = (kc0 - kMot) / kN;
    const bool a = (i == 0) ? false
                 : (i == 1) ? ((d == 0) || (d >= 2))
                            : (d >= 1);
    return a ? 1 : 0;
  }
  const int j = kc0 / kT, cc0 = kc0 % kT, cc1 = cc0 + 31;
  bool blocked, strict;
  if (i == 0)      { blocked = (j != 0); strict = false; }
  else if (i == 1) { blocked = false;    strict = (j != 0); }
  else             { blocked = false;    strict = (j == 2); }
  if (blocked) return 0;
  const bool any = strict ? (cc0 < rr1) : (cc0 <= rr1);
  if (!any) return 0;
  const bool all = strict ? (cc1 < rr0) : (cc1 <= rr0);
  return all ? 1 : 2;
}

// ---------------------------------------------------------------------------
// GEMM: Y[M x 512] = X[M x 512] @ W[512 x 512] + bias
//   in_mode : 0 = X is f32 flat [M][C]; 1 = X is bf16 head-split [B][H][L][dh]
//   out_mode: 0 = Y is bf16 head-split; 1 = Y is f32 flat (final output)
// Block: 256 threads (8 waves as 4 row x 2 col), tile 64x128, K-chunks of 32.
// ---------------------------------------------------------------------------
__global__ __launch_bounds__(256)
void proj_gemm_kernel(const void* __restrict__ Xv, const float* __restrict__ W,
                      const float* __restrict__ bias, void* __restrict__ Yv,
                      int in_mode, int out_mode) {
  __shared__ alignas(16) bf16_t Xs[64][40];    // 64 rows x 32 k (pad->40)
  __shared__ alignas(16) bf16_t Ws[32][136];   // 32 k x 128 cols (pad->136)

  const int tid  = threadIdx.x;
  const int wave = tid >> 5, lane = tid & 31;
  const int hi   = lane >> 4, ln = lane & 15;
  const int wr   = wave & 3,  wc = wave >> 2;
  const int row0 = blockIdx.x * 64;
  const int col0 = blockIdx.y * 128;

  const float*  Xf = (const float*)Xv;
  const bf16_t* Xb = (const bf16_t*)Xv;

  v8f acc[4];
#pragma unroll
  for (int t = 0; t < 4; ++t)
#pragma unroll
    for (int g = 0; g < 8; ++g) acc[t][g] = 0.0f;

  for (int kb = 0; kb < kC / 32; ++kb) {
    // ---- stage X tile (64x32) as bf16 -------------------------------------
    {
      const int r  = tid >> 2;
      const int c8 = (tid & 3) * 8;
      if (in_mode == 0) {
        const float* xp = Xf + (size_t)(row0 + r) * kC + kb * 32 + c8;
        v4f x0 = *(const v4f*)xp;
        v4f x1 = *(const v4f*)(xp + 4);
        v8bf xb;
#pragma unroll
        for (int e = 0; e < 4; ++e) { xb[e] = (bf16_t)x0[e]; xb[e + 4] = (bf16_t)x1[e]; }
        *(v8bf*)&Xs[r][c8] = xb;
      } else {
        // Already bf16 in memory: pure copy -> async DMA into LDS (ASYNCcnt).
        const int m = row0 + r;
        const int b = m / kL, l = m % kL;
        const int c = kb * 32 + c8;
        const int h = c >> 6, d = c & 63;
        const bf16_t* xp = Xb + (((size_t)(b * kH + h) * kL + l) * kDH + d);
        async_load_b128_to_lds(xp, &Xs[r][c8]);
      }
    }
    // ---- stage W tile (32x128) as bf16 ------------------------------------
    {
      const int kk  = tid >> 3;
      const int c16 = (tid & 7) * 16;
      const float* wp = W + (size_t)(kb * 32 + kk) * kC + col0 + c16;
      v4f w0 = *(const v4f*)wp;
      v4f w1 = *(const v4f*)(wp + 4);
      v4f w2 = *(const v4f*)(wp + 8);
      v4f w3 = *(const v4f*)(wp + 12);
      v8bf b0, b1;
#pragma unroll
      for (int e = 0; e < 4; ++e) {
        b0[e] = (bf16_t)w0[e]; b0[e + 4] = (bf16_t)w1[e];
        b1[e] = (bf16_t)w2[e]; b1[e + 4] = (bf16_t)w3[e];
      }
      *(v8bf*)&Ws[kk][c16]     = b0;
      *(v8bf*)&Ws[kk][c16 + 8] = b1;
      if (kb + 1 < kC / 32)   // prefetch next W tile chunk into caches
        __builtin_prefetch(wp + kC * 32, 0, 0);
    }
    wait_asynccnt0();          // async LDS writes visible before barrier
    __syncthreads();

    // A fragment: 16x32, lane holds row M=ln, K groups per ISA layout.
    const bf16_t* xr = &Xs[wr * 16 + ln][0];
    v16bf a = cat8(*(const v8bf*)(xr + 8 * hi),
                   *(const v8bf*)(xr + 16 + 8 * hi));
#pragma unroll
    for (int t = 0; t < 4; ++t) {
      // B fragment: 32x16, lane holds N column = ln, K rows per ISA layout.
      const int wcn = wc * 64 + t * 16 + ln;
      v16bf bb;
#pragma unroll
      for (int e = 0; e < 8; ++e) { bb[e] = Ws[8 * hi + e][wcn]; bb[e + 8] = Ws[16 + 8 * hi + e][wcn]; }
      acc[t] = wmma_bf16(a, bb, acc[t]);
    }
    __syncthreads();
  }

  // ---- write out (C/D layout: VGPR g -> row g+8*hi, col = ln) -------------
  bf16_t* Yb = (bf16_t*)Yv;
  float*  Yf = (float*)Yv;
#pragma unroll
  for (int t = 0; t < 4; ++t) {
    const int col = col0 + wc * 64 + t * 16 + ln;
    const float bcol = bias[col];
#pragma unroll
    for (int g = 0; g < 8; ++g) {
      const int m = row0 + wr * 16 + g + 8 * hi;
      const float v = acc[t][g] + bcol;
      if (out_mode == 0) {
        const int b = m / kL, l = m % kL;
        const int h = col >> 6, d = col & 63;
        Yb[((size_t)(b * kH + h) * kL + l) * kDH + d] = (bf16_t)v;
      } else {
        Yf[(size_t)m * kC + col] = v;
      }
    }
  }
}

// ---------------------------------------------------------------------------
// Flash attention: 4 waves/block, one 16-row query tile per wave, block shares
// (b,h) so K/V tiles (32 keys) are staged cooperatively in LDS.
// V is staged via async global->LDS DMA; K is staged transposed through VGPRs.
// ---------------------------------------------------------------------------
__global__ __launch_bounds__(128)
void attn_kernel(const bf16_t* __restrict__ Q, const bf16_t* __restrict__ K,
                 const bf16_t* __restrict__ V, bf16_t* __restrict__ O) {
  __shared__ alignas(16) bf16_t Ks[kDH][40];    // transposed: [d][key]
  __shared__ alignas(16) bf16_t Vs[32][72];     // [key][d]
  __shared__ alignas(16) bf16_t Ps[4][16][40];  // per-wave P tile [q][key]

  const int tid  = threadIdx.x;
  const int wave = tid >> 5, lane = tid & 31;
  const int hi   = lane >> 4, ln = lane & 15;
  const int bh   = blockIdx.y;
  const int qr0  = (blockIdx.x * 4 + wave) * 16;

  const bf16_t* Qb = Q + (size_t)bh * kL * kDH;
  const bf16_t* Kb = K + (size_t)bh * kL * kDH;
  const bf16_t* Vb = V + (size_t)bh * kL * kDH;
  bf16_t*       Ob = O + (size_t)bh * kL * kDH;

  const float NEG_INF = -__builtin_inff();

  // Q A-fragments for d-chunks [0,32) and [32,64).
  v16bf aq[2];
  {
    const bf16_t* qp = Qb + (size_t)(qr0 + ln) * kDH;
#pragma unroll
    for (int dc = 0; dc < 2; ++dc)
      aq[dc] = cat8(*(const v8bf*)(qp + dc * 32 + 8 * hi),
                    *(const v8bf*)(qp + dc * 32 + 16 + 8 * hi));
  }

  float m_run[8], l_run[8];
  v8f oacc[4];
#pragma unroll
  for (int g = 0; g < 8; ++g) { m_run[g] = NEG_INF; l_run[g] = 0.0f; }
#pragma unroll
  for (int t = 0; t < 4; ++t)
#pragma unroll
    for (int g = 0; g < 8; ++g) oacc[t][g] = 0.0f;

  for (int kc0 = 0; kc0 < kL; kc0 += 32) {
    // ---- cooperative staging: V via async DMA, K transposed via VGPRs -----
    {
      const int key = tid >> 2;
      const int d0  = (tid & 3) * 16;
      // V: natural [key][d] layout -> pure copy, ASYNCcnt-tracked DMA.
      const bf16_t* vp = Vb + (size_t)(kc0 + key) * kDH + d0;
      async_load_b128_to_lds(vp,     &Vs[key][d0]);
      async_load_b128_to_lds(vp + 8, &Vs[key][d0 + 8]);
      // K: needs [d][key] transpose -> load to VGPRs, scatter-store to LDS.
      const bf16_t* kp = Kb + (size_t)(kc0 + key) * kDH + d0;
      v8bf k0 = *(const v8bf*)kp;
      v8bf k1 = *(const v8bf*)(kp + 8);
#pragma unroll
      for (int e = 0; e < 8; ++e) {
        Ks[d0 + e][key]     = k0[e];
        Ks[d0 + 8 + e][key] = k1[e];
      }
      if (kc0 + 32 < kL) {   // prefetch next tile's K/V lines
        __builtin_prefetch(kp + 32 * kDH, 0, 0);
        __builtin_prefetch(vp + 32 * kDH, 0, 0);
      }
    }
    wait_asynccnt0();        // async V writes visible before barrier
    __syncthreads();

    const int st = tile_status(qr0, kc0);   // wave-uniform -> EXEC stays full
    if (st != 0) {
      // ---- S = (Q K^T) * scale : 2 key-halves x 2 d-chunks = 4 WMMAs ------
      v8f s[2];
#pragma unroll
      for (int t = 0; t < 2; ++t) {
#pragma unroll
        for (int g = 0; g < 8; ++g) s[t][g] = 0.0f;
#pragma unroll
        for (int dc = 0; dc < 2; ++dc) {
          v16bf bb;
#pragma unroll
          for (int e = 0; e < 8; ++e) { bb[e] = Ks[dc * 32 + 8 * hi + e][t * 16 + ln]; bb[e + 8] = Ks[dc * 32 + 16 + 8 * hi + e][t * 16 + ln]; }
          s[t] = wmma_bf16(aq[dc], bb, s[t]);
        }
      }
      // ---- scale + mask + row max -----------------------------------------
      const float scale = 0.125f;   // 1/sqrt(64)
      float tm[8];
#pragma unroll
      for (int g = 0; g < 8; ++g) tm[g] = NEG_INF;
#pragma unroll
      for (int t = 0; t < 2; ++t)
#pragma unroll
        for (int g = 0; g < 8; ++g) {
          float vv = s[t][g] * scale;
          if (st == 2 && !allowed_rc(qr0 + g + 8 * hi, kc0 + t * 16 + ln))
            vv = NEG_INF;
          s[t][g] = vv;
          tm[g] = fmaxf(tm[g], vv);
        }
#pragma unroll
      for (int off = 1; off < 16; off <<= 1)
#pragma unroll
        for (int g = 0; g < 8; ++g)
          tm[g] = fmaxf(tm[g], __shfl_xor(tm[g], off, 32));
      // ---- online softmax update ------------------------------------------
      float alpha[8];
#pragma unroll
      for (int g = 0; g < 8; ++g) {
        const float mn = fmaxf(m_run[g], tm[g]);
        alpha[g] = (mn == NEG_INF) ? 1.0f : __expf(m_run[g] - mn);
        m_run[g] = mn;
      }
      float rs[8];
#pragma unroll
      for (int g = 0; g < 8; ++g) rs[g] = 0.0f;
#pragma unroll
      for (int t = 0; t < 2; ++t)
#pragma unroll
        for (int g = 0; g < 8; ++g) {
          const float mn = m_run[g];
          const float p  = (mn == NEG_INF) ? 0.0f : __expf(s[t][g] - mn);
          rs[g] += p;
          Ps[wave][g + 8 * hi][t * 16 + ln] = (bf16_t)p;
        }
#pragma unroll
      for (int off = 1; off < 16; off <<= 1)
#pragma unroll
        for (int g = 0; g < 8; ++g) rs[g] += __shfl_xor(rs[g], off, 32);
#pragma unroll
      for (int g = 0; g < 8; ++g) l_run[g] = l_run[g] * alpha[g] + rs[g];
#pragma unroll
      for (int t = 0; t < 4; ++t)
#pragma unroll
        for (int g = 0; g < 8; ++g) oacc[t][g] *= alpha[g];
      // ---- O += P x V : read P back as A-fragment, 4 WMMAs ----------------
      const bf16_t* pr = &Ps[wave][ln][0];
      v16bf ap = cat8(*(const v8bf*)(pr + 8 * hi),
                      *(const v8bf*)(pr + 16 + 8 * hi));
#pragma unroll
      for (int t = 0; t < 4; ++t) {
        v16bf bb;
#pragma unroll
        for (int e = 0; e < 8; ++e) { bb[e] = Vs[8 * hi + e][t * 16 + ln]; bb[e + 8] = Vs[16 + 8 * hi + e][t * 16 + ln]; }
        oacc[t] = wmma_bf16(ap, bb, oacc[t]);
      }
    }
    __syncthreads();
  }

  // ---- normalize and store head-split bf16 --------------------------------
#pragma unroll
  for (int g = 0; g < 8; ++g) {
    const float inv = (l_run[g] > 0.0f) ? (1.0f / l_run[g]) : 0.0f;
    const int r = qr0 + g + 8 * hi;
#pragma unroll
    for (int t = 0; t < 4; ++t)
      Ob[(size_t)r * kDH + t * 16 + ln] = (bf16_t)(oacc[t][g] * inv);
  }
}

// ---------------------------------------------------------------------------
extern "C" void kernel_launch(void* const* d_in, const int* in_sizes, int n_in,
                              void* d_out, int out_size, void* d_ws, size_t ws_size,
                              hipStream_t stream) {
  (void)in_sizes; (void)n_in; (void)out_size; (void)ws_size;
  const float* x  = (const float*)d_in[0];
  const float* Wq = (const float*)d_in[1];
  const float* bq = (const float*)d_in[2];
  const float* Wk = (const float*)d_in[3];
  const float* bk = (const float*)d_in[4];
  const float* Wv = (const float*)d_in[5];
  const float* bv = (const float*)d_in[6];
  const float* Wp = (const float*)d_in[7];
  const float* bp = (const float*)d_in[8];

  const size_t S = (size_t)kB * kH * kL * kDH;   // elements per bf16 tensor
  bf16_t* q = (bf16_t*)d_ws;
  bf16_t* k = q + S;
  bf16_t* v = k + S;
  bf16_t* o = v + S;

  dim3 gemm_grid(kM / 64, kC / 128);   // (106, 4)
  proj_gemm_kernel<<<gemm_grid, 256, 0, stream>>>(x, Wq, bq, q, 0, 0);
  proj_gemm_kernel<<<gemm_grid, 256, 0, stream>>>(x, Wk, bk, k, 0, 0);
  proj_gemm_kernel<<<gemm_grid, 256, 0, stream>>>(x, Wv, bv, v, 0, 0);

  attn_kernel<<<dim3(kL / 16 / 4, kB * kH), 128, 0, stream>>>(q, k, v, o);

  proj_gemm_kernel<<<gemm_grid, 256, 0, stream>>>(o, Wp, bp, d_out, 1, 1);
}
